// LinearSelfAttentionBlock_75041668596017
// MI455X (gfx1250) — matrix operand order repaired and checked by reading, not compiled, not run on hardware
//
#include <hip/hip_runtime.h>

// ---------------------------------------------------------------------------
// LinearSelfAttentionBlock, chunked causal linear-attention scan for gfx1250.
// B=4, N=8192, H=T=128, chunk C=128, 64 chunks/batch, 256 chunks total.
// bf16 WMMA (v_wmma_f32_16x16x32_bf16) with f32 accumulation.
// ---------------------------------------------------------------------------

typedef __attribute__((ext_vector_type(16))) __bf16 v16bf;
typedef __attribute__((ext_vector_type(8)))  float  v8f;

#define SF 132   // f32 LDS row stride (128 + 4 pad)
#define SB 136   // bf16 LDS row stride (128 + 8 pad)

__device__ __forceinline__ v8f vzero8() {
  v8f z;
#pragma unroll
  for (int i = 0; i < 8; ++i) z[i] = 0.0f;
  return z;
}

__device__ __forceinline__ v8f wmma_bf16(v16bf a, v16bf b, v8f c) {
  return __builtin_amdgcn_wmma_f32_16x16x32_bf16(false, a, false, b, (short)0, c,
                                                 false, false);
}

// A operand (16x32 bf16) from row-major [m][stride] storage.
// lane<16: row m0+lane, K = k0+0..15
// lane>=16: row m0+lane-16, K = k0+16..31
__device__ __forceinline__ v16bf ldA_row(const __bf16* base, int stride, int m0,
                                         int k0, int lane) {
  const __bf16* p = base + (m0 + (lane & 15)) * stride + k0 + ((lane & 16) ? 16 : 0);
  v16bf r;
#pragma unroll
  for (int e = 0; e < 8; ++e) r[e] = p[e];
#pragma unroll
  for (int e = 0; e < 8; ++e) r[8 + e] = p[8 + e];
  return r;
}

// B operand (32x16 bf16): logical B[k][n] stored as base[n*stride + k]
// (row n contiguous in k). lane<16: col n0+lane, K=k0..k0+15;
// lane>=16: col n0+lane-16, K=k0+16..k0+31.
__device__ __forceinline__ v16bf ldB_row(const __bf16* base, int stride, int k0,
                                         int n0, int lane) {
  const __bf16* p = base + (n0 + (lane & 15)) * stride + k0 + ((lane & 16) ? 16 : 0);
  v16bf r;
#pragma unroll
  for (int e = 0; e < 16; ++e) r[e] = p[e];
  return r;
}

__device__ __forceinline__ v16bf ldB_row_f32(const float* base, int stride, int k0,
                                             int n0, int lane) {
  const float* p = base + (n0 + (lane & 15)) * stride + k0 + ((lane & 16) ? 16 : 0);
  v16bf r;
#pragma unroll
  for (int e = 0; e < 16; ++e) r[e] = (__bf16)p[e];
  return r;
}

// D tile (f32 16x16): VGPR r -> M = m0 + r + (lane>=16 ? 8:0), N = n0 + (lane&15)
__device__ __forceinline__ void stD_rowbf(__bf16* base, int stride, int m0, int n0,
                                          v8f acc, int lane) {
  int n = n0 + (lane & 15);
  int m = m0 + ((lane & 16) ? 8 : 0);
#pragma unroll
  for (int r = 0; r < 8; ++r) base[(m + r) * stride + n] = (__bf16)acc[r];
}

__device__ __forceinline__ void stD_colbf(__bf16* base, int stride, int m0, int n0,
                                          v8f acc, int lane) {
  int n = n0 + (lane & 15);
  int m = m0 + ((lane & 16) ? 8 : 0);
  __bf16* p = base + n * stride + m;
#pragma unroll
  for (int r = 0; r < 8; ++r) p[r] = (__bf16)acc[r];
}

// ---------------------------------------------------------------------------
// Kernel 0: f32 -> bf16 weight conversion (4 matrices of 128x128)
// ---------------------------------------------------------------------------
__global__ void __launch_bounds__(256) k_cvt_weights(
    const float* __restrict__ w0, const float* __restrict__ w1,
    const float* __restrict__ w2, const float* __restrict__ w3,
    __bf16* __restrict__ o0, __bf16* __restrict__ o1,
    __bf16* __restrict__ o2, __bf16* __restrict__ o3) {
  int i = blockIdx.x * 256 + threadIdx.x;  // 0..16383
  const float* s;
  __bf16* d;
  switch (blockIdx.y) {
    case 0: s = w0; d = o0; break;
    case 1: s = w1; d = o1; break;
    case 2: s = w2; d = o2; break;
    default: s = w3; d = o3; break;
  }
  d[i] = (__bf16)s[i];
}

// ---------------------------------------------------------------------------
// Kernel 1: per chunk -> LN1, K & V projections, chunk state S = K^T V
//   Kbuf: bf16 [chunk][m][h] row-major   Vbuf: bf16 [chunk][v][m] transposed
//   Sbuf: f32  [chunk][v][h] transposed (chunk sums; prefixed later)
// ---------------------------------------------------------------------------
__global__ void __launch_bounds__(256) k_proj_state(
    const float* __restrict__ tokens, const __bf16* __restrict__ Wk,
    const __bf16* __restrict__ Wv, const float* __restrict__ g1,
    const float* __restrict__ b1, __bf16* __restrict__ Kbuf,
    __bf16* __restrict__ Vbuf, float* __restrict__ Sbuf) {
  __shared__ __align__(16) unsigned char smem[105472];
  float* tokf = (float*)smem;                      // [128][SF] (dies after LN)
  __bf16* KT = (__bf16*)smem;                      // [128][SB] KT[h][m]
  __bf16* VT = (__bf16*)(smem + 34816);            // [128][SB] VT[v][m]
  __bf16* tnb = (__bf16*)(smem + 69632);           // [128][SB] tn bf16
  float* mu = (float*)(smem + 104448);
  float* rs = mu + 128;

  const int bc = blockIdx.x;  // batch*64 + chunk
  const int t = threadIdx.x, lane = t & 31, wave = t >> 5;
  const int m0 = wave * 16;
  const float* tok = tokens + (size_t)bc * 16384;

  // stage 1: stage tokens into LDS (coalesced)
  for (int i = t; i < 16384; i += 256) tokf[(i >> 7) * SF + (i & 127)] = tok[i];
  __syncthreads();

  // stage 2: LN1 statistics (one thread per row)
  if (t < 128) {
    const float* row = tokf + t * SF;
    float s = 0.f, ss = 0.f;
    for (int k = 0; k < 128; ++k) { float x = row[k]; s += x; ss += x * x; }
    float m = s * (1.f / 128.f);
    mu[t] = m;
    rs[t] = rsqrtf(ss * (1.f / 128.f) - m * m + 1e-5f);
  }
  __syncthreads();

  // stage 3: normalized tokens -> bf16
  for (int i = t; i < 16384; i += 256) {
    int r = i >> 7, k = i & 127;
    float x = (tokf[r * SF + k] - mu[r]) * rs[r] * g1[k] + b1[k];
    tnb[r * SB + k] = (__bf16)x;
  }
  __syncthreads();  // tokf dead; KT/VT region becomes live

  __bf16* Kc = Kbuf + (size_t)bc * 16384;
  __bf16* Vc = Vbuf + (size_t)bc * 16384;

  // stage 4a: K = tn @ Wk^T  (store KT in LDS + K row-major in global)
  {
    v8f acc[8];
#pragma unroll
    for (int i = 0; i < 8; ++i) acc[i] = vzero8();
#pragma unroll
    for (int ks = 0; ks < 4; ++ks) {
      v16bf a = ldA_row(tnb, SB, m0, ks * 32, lane);
#pragma unroll
      for (int nt = 0; nt < 8; ++nt) {
        v16bf b = ldB_row(Wk, 128, ks * 32, nt * 16, lane);
        acc[nt] = wmma_bf16(a, b, acc[nt]);
      }
    }
#pragma unroll
    for (int nt = 0; nt < 8; ++nt) {
      stD_colbf(KT, SB, m0, nt * 16, acc[nt], lane);
      int n = nt * 16 + (lane & 15);
      int mm = m0 + ((lane & 16) ? 8 : 0);
#pragma unroll
      for (int r = 0; r < 8; ++r) Kc[(mm + r) * 128 + n] = (__bf16)acc[nt][r];
    }
  }
  // stage 4b: V = tn @ Wv^T  (store VT in LDS + V transposed in global)
  {
    v8f acc[8];
#pragma unroll
    for (int i = 0; i < 8; ++i) acc[i] = vzero8();
#pragma unroll
    for (int ks = 0; ks < 4; ++ks) {
      v16bf a = ldA_row(tnb, SB, m0, ks * 32, lane);
#pragma unroll
      for (int nt = 0; nt < 8; ++nt) {
        v16bf b = ldB_row(Wv, 128, ks * 32, nt * 16, lane);
        acc[nt] = wmma_bf16(a, b, acc[nt]);
      }
    }
#pragma unroll
    for (int nt = 0; nt < 8; ++nt) {
      stD_colbf(VT, SB, m0, nt * 16, acc[nt], lane);
      int n = nt * 16 + (lane & 15);
      int mm = m0 + ((lane & 16) ? 8 : 0);
      __bf16* p = Vc + (size_t)n * 128 + mm;
#pragma unroll
      for (int r = 0; r < 8; ++r) p[r] = (__bf16)acc[nt][r];
    }
  }
  __syncthreads();  // KT/VT fully written by all waves

  // stage 5: S = K^T @ V  (M = h strip, N = v, K-dim = m)
  {
    float* Sc = Sbuf + (size_t)bc * 16384;
    v8f acc[8];
#pragma unroll
    for (int i = 0; i < 8; ++i) acc[i] = vzero8();
#pragma unroll
    for (int ks = 0; ks < 4; ++ks) {
      v16bf a = ldA_row(KT, SB, m0, ks * 32, lane);  // rows h, contiguous m
#pragma unroll
      for (int vt = 0; vt < 8; ++vt) {
        v16bf b = ldB_row(VT, SB, ks * 32, vt * 16, lane);  // (k=m, n=v)
        acc[vt] = wmma_bf16(a, b, acc[vt]);
      }
    }
#pragma unroll
    for (int vt = 0; vt < 8; ++vt) {  // transposed store: Sc[v*128 + h]
      int n = vt * 16 + (lane & 15);
      int mm = m0 + ((lane & 16) ? 8 : 0);
      float* p = Sc + (size_t)n * 128 + mm;
#pragma unroll
      for (int r = 0; r < 8; ++r) p[r] = acc[vt][r];
    }
  }
}

// ---------------------------------------------------------------------------
// Kernel 2: exclusive prefix sum of chunk states along the 64 chunks per batch
// ---------------------------------------------------------------------------
__global__ void __launch_bounds__(256) k_prefix(float* __restrict__ S) {
  int b = blockIdx.x >> 4, sl = blockIdx.x & 15;
  for (int i = 0; i < 4; ++i) {
    int idx = sl * 1024 + i * 256 + threadIdx.x;  // 0..16383
    float* p = S + (size_t)b * 64 * 16384 + idx;
    float run = 0.f;
    for (int c = 0; c < 64; ++c) {
      float v = p[(size_t)c * 16384];
      p[(size_t)c * 16384] = run;
      run += v;
    }
  }
}

// ---------------------------------------------------------------------------
// Kernel 3: per chunk -> Q, causal intra scores, attn = scores@V + Q@S_prefix,
//           output projection, residual, LN2, final store.
// ---------------------------------------------------------------------------
__global__ void __launch_bounds__(256) k_attn_out(
    const float* __restrict__ tokens, const __bf16* __restrict__ Wq,
    const __bf16* __restrict__ Wo, const float* __restrict__ g1,
    const float* __restrict__ b1, const float* __restrict__ g2,
    const float* __restrict__ b2, const __bf16* __restrict__ Kbuf,
    const __bf16* __restrict__ Vbuf, const float* __restrict__ Sbuf,
    float* __restrict__ out) {
  __shared__ __align__(16) unsigned char smem[103424];
  float* tokf = (float*)smem;                     // [128][SF] tokens -> out_pre
  __bf16* work = (__bf16*)(smem + 67584);         // [128][SB] tn -> Q -> scores -> attn
  float* mu = (float*)(smem + 102400);
  float* rs = mu + 128;

  const int bc = blockIdx.x;
  const int t = threadIdx.x, lane = t & 31, wave = t >> 5;
  const int m0 = wave * 16;
  const float* tok = tokens + (size_t)bc * 16384;
  const __bf16* Kc = Kbuf + (size_t)bc * 16384;  // [j][h]
  const __bf16* Vc = Vbuf + (size_t)bc * 16384;  // [v][m]
  const float* Sc = Sbuf + (size_t)bc * 16384;   // [v][h] exclusive prefix

  // stage 1: tokens -> LDS
  for (int i = t; i < 16384; i += 256) tokf[(i >> 7) * SF + (i & 127)] = tok[i];
  __syncthreads();
  // stage 2: LN1 stats
  if (t < 128) {
    const float* row = tokf + t * SF;
    float s = 0.f, ss = 0.f;
    for (int k = 0; k < 128; ++k) { float x = row[k]; s += x; ss += x * x; }
    float m = s * (1.f / 128.f);
    mu[t] = m;
    rs[t] = rsqrtf(ss * (1.f / 128.f) - m * m + 1e-5f);
  }
  __syncthreads();
  // stage 3: tn -> work (bf16)
  for (int i = t; i < 16384; i += 256) {
    int r = i >> 7, k = i & 127;
    float x = (tokf[r * SF + k] - mu[r]) * rs[r] * g1[k] + b1[k];
    work[r * SB + k] = (__bf16)x;
  }
  __syncthreads();

  // stage 4: Qs = (tn @ Wq^T) / sqrt(128), written back into this wave's strip
  {
    v8f acc[8];
#pragma unroll
    for (int i = 0; i < 8; ++i) acc[i] = vzero8();
#pragma unroll
    for (int ks = 0; ks < 4; ++ks) {
      v16bf a = ldA_row(work, SB, m0, ks * 32, lane);
#pragma unroll
      for (int nt = 0; nt < 8; ++nt) {
        v16bf b = ldB_row(Wq, 128, ks * 32, nt * 16, lane);
        acc[nt] = wmma_bf16(a, b, acc[nt]);
      }
    }
    const float qs = 0.08838834764831845f;  // 1/sqrt(128)
#pragma unroll
    for (int nt = 0; nt < 8; ++nt) {
      int n = nt * 16 + (lane & 15);
      int mm = m0 + ((lane & 16) ? 8 : 0);
#pragma unroll
      for (int r = 0; r < 8; ++r) work[(mm + r) * SB + n] = (__bf16)(acc[nt][r] * qs);
    }
  }

  // capture Q strip as A fragments (kept in registers for scores + inter)
  v16bf aQ[4];
#pragma unroll
  for (int ks = 0; ks < 4; ++ks) aQ[ks] = ldA_row(work, SB, m0, ks * 32, lane);

  // stage 5: causal intra-chunk scores into the same strip rows.
  // zero masked tiles (jt > wave) so the attn K-loop can run full width
  for (int jt = wave + 1; jt < 8; ++jt) {
    int n = jt * 16 + (lane & 15);
    int mm = m0 + ((lane & 16) ? 8 : 0);
#pragma unroll
    for (int r = 0; r < 8; ++r) work[(mm + r) * SB + n] = (__bf16)0.f;
  }
  for (int jt = 0; jt <= wave; ++jt) {
    v8f sa = vzero8();
#pragma unroll
    for (int ks = 0; ks < 4; ++ks) {
      v16bf b = ldB_row(Kc, 128, ks * 32, jt * 16, lane);  // (k=h, n=j) = K[j][h]
      sa = wmma_bf16(aQ[ks], b, sa);
    }
    int n = jt * 16 + (lane & 15);
    int mm = m0 + ((lane & 16) ? 8 : 0);
    if (jt == wave) {  // diagonal tile: keep j <= i
#pragma unroll
      for (int r = 0; r < 8; ++r)
        work[(mm + r) * SB + n] = (__bf16)((n <= mm + r) ? sa[r] : 0.f);
    } else {
#pragma unroll
      for (int r = 0; r < 8; ++r) work[(mm + r) * SB + n] = (__bf16)sa[r];
    }
  }

  // stage 6: attn = scores @ V  +  Qs @ S_prefix
  v8f at[8];
#pragma unroll
  for (int i = 0; i < 8; ++i) at[i] = vzero8();
#pragma unroll
  for (int ks = 0; ks < 4; ++ks) {
    v16bf a = ldA_row(work, SB, m0, ks * 32, lane);  // scores strip
#pragma unroll
    for (int vt = 0; vt < 8; ++vt) {
      v16bf b = ldB_row(Vc, 128, ks * 32, vt * 16, lane);  // (k=j, n=v) = V[j][v]
      at[vt] = wmma_bf16(a, b, at[vt]);
    }
  }
#pragma unroll
  for (int ks = 0; ks < 4; ++ks) {
#pragma unroll
    for (int vt = 0; vt < 8; ++vt) {
      v16bf b = ldB_row_f32(Sc, 128, ks * 32, vt * 16, lane);  // (k=h, n=v)
      at[vt] = wmma_bf16(aQ[ks], b, at[vt]);
    }
  }
#pragma unroll
  for (int vt = 0; vt < 8; ++vt) stD_rowbf(work, SB, m0, vt * 16, at[vt], lane);

  // stage 7: out = attn @ Wo^T * 0.1 + tokens (into tokf)
  {
    v8f oa[8];
#pragma unroll
    for (int i = 0; i < 8; ++i) oa[i] = vzero8();
#pragma unroll
    for (int ks = 0; ks < 4; ++ks) {
      v16bf a = ldA_row(work, SB, m0, ks * 32, lane);  // attn strip
#pragma unroll
      for (int ot = 0; ot < 8; ++ot) {
        v16bf b = ldB_row(Wo, 128, ks * 32, ot * 16, lane);
        oa[ot] = wmma_bf16(a, b, oa[ot]);
      }
    }
#pragma unroll
    for (int ot = 0; ot < 8; ++ot) {
      int n = ot * 16 + (lane & 15);
      int mm = m0 + ((lane & 16) ? 8 : 0);
#pragma unroll
      for (int r = 0; r < 8; ++r) {
        float* p = &tokf[(mm + r) * SF + n];
        *p = *p + 0.1f * oa[ot][r];
      }
    }
  }
  __syncthreads();

  // stage 8: LN2 + store
  if (t < 128) {
    const float* row = tokf + t * SF;
    float s = 0.f, ss = 0.f;
    for (int k = 0; k < 128; ++k) { float x = row[k]; s += x; ss += x * x; }
    float m = s * (1.f / 128.f);
    mu[t] = m;
    rs[t] = rsqrtf(ss * (1.f / 128.f) - m * m + 1e-5f);
  }
  __syncthreads();
  float* og = out + (size_t)bc * 16384;
  for (int i = t; i < 16384; i += 256) {
    int r = i >> 7, k = i & 127;
    og[i] = (tokf[r * SF + k] - mu[r]) * rs[r] * g2[k] + b2[k];
  }
}

// ---------------------------------------------------------------------------
extern "C" void kernel_launch(void* const* d_in, const int* in_sizes, int n_in,
                              void* d_out, int out_size, void* d_ws, size_t ws_size,
                              hipStream_t stream) {
  const float* tokens = (const float*)d_in[0];
  const float* Wq = (const float*)d_in[1];
  const float* Wk = (const float*)d_in[2];
  const float* Wv = (const float*)d_in[3];
  const float* Wo = (const float*)d_in[4];
  const float* g1 = (const float*)d_in[5];
  const float* b1 = (const float*)d_in[6];
  const float* g2 = (const float*)d_in[7];
  const float* b2 = (const float*)d_in[8];
  float* out = (float*)d_out;

  // workspace layout
  char* ws = (char*)d_ws;
  __bf16* Wqb = (__bf16*)ws;              // 16384 elems
  __bf16* Wkb = Wqb + 16384;
  __bf16* Wvb = Wkb + 16384;
  __bf16* Wob = Wvb + 16384;
  __bf16* Kbuf = (__bf16*)(ws + 131072);              // 4M bf16 (8 MB)
  __bf16* Vbuf = Kbuf + (size_t)4 * 1024 * 1024;      // 4M bf16 (8 MB)
  float* Sbuf = (float*)(ws + 131072 + 2 * 8388608);  // 4M f32 (16 MB)

  k_cvt_weights<<<dim3(64, 4), 256, 0, stream>>>(Wq, Wk, Wv, Wo, Wqb, Wkb, Wvb, Wob);
  k_proj_state<<<256, 256, 0, stream>>>(tokens, Wkb, Wvb, g1, b1, Kbuf, Vbuf, Sbuf);
  k_prefix<<<64, 256, 0, stream>>>(Sbuf);
  k_attn_out<<<256, 256, 0, stream>>>(tokens, Wqb, Wob, g1, b1, g2, b2, Kbuf, Vbuf,
                                      Sbuf, out);
}
